// Qwen3MoeTritonSparseMoeBlock_76647986365149
// MI455X (gfx1250) — matrix operand order repaired and checked by reading, not compiled, not run on hardware
//
#include <hip/hip_runtime.h>
#include <hip/hip_bf16.h>
#include <math.h>

#define T_TOK 4096
#define H_DIM 1024
#define E_NUM 16
#define TOPK  2
#define I_DIM 768
#define NPAIR (T_TOK * TOPK)   // 8192

typedef __attribute__((ext_vector_type(16))) __bf16 v16bf;
typedef __attribute__((ext_vector_type(8)))  float  v8f;

union AFrag {
  uint4        q[2];
  unsigned int u[8];
  v16bf        v;
};

__device__ __forceinline__ unsigned short f2bf(float f) {
  unsigned int u = __builtin_bit_cast(unsigned int, f);
  u += 0x7FFFu + ((u >> 16) & 1u);          // round-to-nearest-even
  return (unsigned short)(u >> 16);
}

// ---------------------------------------------------------------- meta init
__global__ void init_meta_kernel(int* __restrict__ cnt) {
  if (threadIdx.x < E_NUM) cnt[threadIdx.x] = 0;
}

// ---------------------------------------------------------------- x -> bf16
__global__ void cvt_x_kernel(const float* __restrict__ x,
                             unsigned short* __restrict__ xb) {
  int i = blockIdx.x * blockDim.x + threadIdx.x;
  xb[i] = f2bf(x[i]);
}

// ---------------------------------------------------------------- zero out
__global__ void zero_kernel(float* __restrict__ out) {
  out[blockIdx.x * blockDim.x + threadIdx.x] = 0.f;
}

// ---------------------------------------------------------------- weight convert+transpose
// in : slab e of (Kdim, Ndim) fp32, row-major
// out: slab e of (Ndim, Kdim) bf16  (column of in -> contiguous row of out)
__global__ __launch_bounds__(256)
void cvt_transpose_kernel(const float* __restrict__ in,
                          unsigned short* __restrict__ outT,
                          int Kdim, int Ndim) {
  __shared__ unsigned short tile[32][33];
  const int k0 = blockIdx.x * 32;
  const int n0 = blockIdx.y * 32;
  const int e  = blockIdx.z;
  const float*    ine  = in   + (size_t)e * Kdim * Ndim;
  unsigned short* oute = outT + (size_t)e * Kdim * Ndim;
  const int tx = threadIdx.x & 31;
  const int ty = threadIdx.x >> 5;           // 0..7
#pragma unroll
  for (int i = 0; i < 4; ++i) {
    const int k = k0 + ty + 8 * i;
    tile[ty + 8 * i][tx] = f2bf(ine[(size_t)k * Ndim + n0 + tx]);
  }
  __syncthreads();
  const int c  = threadIdx.x & 15;           // uint (k-pair) within row
  const int n8 = threadIdx.x >> 4;           // 0..15
#pragma unroll
  for (int i = 0; i < 2; ++i) {
    const int np = n8 + 16 * i;              // 0..31
    const unsigned int v = (unsigned int)tile[2 * c][np] |
                           ((unsigned int)tile[2 * c + 1][np] << 16);
    *(unsigned int*)(oute + (size_t)(n0 + np) * Kdim + k0 + 2 * c) = v;
  }
}

// ---------------------------------------------------------------- router
// one wave32 per token; 8 tokens per 256-thread block
__global__ __launch_bounds__(256)
void gate_topk_kernel(const float* __restrict__ x,
                      const float* __restrict__ gw,
                      int*   __restrict__ topk_idx,
                      float* __restrict__ topk_w,
                      int*   __restrict__ cnt) {
  const int tid  = threadIdx.x;
  const int lane = tid & 31;
  const int wave = tid >> 5;
  const int t    = blockIdx.x * 8 + wave;
  if (t >= T_TOK) return;

  float acc[E_NUM];
#pragma unroll
  for (int e = 0; e < E_NUM; ++e) acc[e] = 0.f;

  const float* xr = x + (size_t)t * H_DIM;
  for (int j = 0; j < H_DIM / 32; ++j) {
    const int hidx = lane + 32 * j;
    const float xv = xr[hidx];
    const float4* g4 = (const float4*)(gw + (size_t)hidx * E_NUM);
    float4 a = g4[0], b = g4[1], c = g4[2], d = g4[3];
    acc[0]  += xv * a.x; acc[1]  += xv * a.y; acc[2]  += xv * a.z; acc[3]  += xv * a.w;
    acc[4]  += xv * b.x; acc[5]  += xv * b.y; acc[6]  += xv * b.z; acc[7]  += xv * b.w;
    acc[8]  += xv * c.x; acc[9]  += xv * c.y; acc[10] += xv * c.z; acc[11] += xv * c.w;
    acc[12] += xv * d.x; acc[13] += xv * d.y; acc[14] += xv * d.z; acc[15] += xv * d.w;
  }
#pragma unroll
  for (int e = 0; e < E_NUM; ++e)
    for (int s = 16; s > 0; s >>= 1)
      acc[e] += __shfl_xor(acc[e], s, 32);

  if (lane == 0) {
    int   i0 = 0;  float b0 = acc[0];
    for (int e = 1; e < E_NUM; ++e) if (acc[e] > b0) { b0 = acc[e]; i0 = e; }
    int   i1 = -1; float b1 = -3.4e38f;
    for (int e = 0; e < E_NUM; ++e) if (e != i0 && acc[e] > b1) { b1 = acc[e]; i1 = e; }
    // softmax denominator cancels under top-k renormalization:
    const float w0 = 1.f / (1.f + __expf(b1 - b0));
    const float w1 = 1.f - w0;
    topk_idx[2 * t] = i0; topk_idx[2 * t + 1] = i1;
    topk_w [2 * t] = w0; topk_w [2 * t + 1] = w1;
    atomicAdd(&cnt[i0], 1);
    atomicAdd(&cnt[i1], 1);
  }
}

// ---------------------------------------------------------------- scan
__global__ void scan_kernel(const int* __restrict__ cnt,
                            int* __restrict__ offs,
                            int* __restrict__ cursor) {
  if (threadIdx.x == 0) {
    int s = 0;
    for (int e = 0; e < E_NUM; ++e) { offs[e] = s; s += cnt[e]; }
    offs[E_NUM] = s;
  }
  if (threadIdx.x < E_NUM) cursor[threadIdx.x] = 0;
}

// ---------------------------------------------------------------- pair lists
__global__ void build_kernel(const int* __restrict__ topk_idx,
                             const float* __restrict__ topk_w,
                             const int* __restrict__ offs,
                             int*   __restrict__ cursor,
                             int*   __restrict__ pair_tok,
                             float* __restrict__ pair_w) {
  const int t = blockIdx.x * blockDim.x + threadIdx.x;
  if (t >= T_TOK) return;
  for (int s = 0; s < TOPK; ++s) {
    const int e = topk_idx[2 * t + s];
    const int pos = atomicAdd(&cursor[e], 1);
    const int d = offs[e] + pos;
    pair_tok[d] = t;
    pair_w[d]   = topk_w[2 * t + s];
  }
}

// ---------------------------------------------------------------- GEMM1
// h = silu(x@Wg) * (x@Wu); tile 16 pairs x 128 cols of I (8 waves x 16 cols)
// A: xb rows (bf16, K-contiguous); B: gupT (E,2I,H) bf16, K-contiguous per column
__global__ __launch_bounds__(256)
void gemm1_kernel(const unsigned short* __restrict__ xb,
                  const unsigned short* __restrict__ gupT,
                  const int* __restrict__ offs,
                  const int* __restrict__ pair_tok,
                  unsigned short* __restrict__ hbuf) {
  const int e    = blockIdx.x;
  const int tile = blockIdx.y;
  const int nz   = blockIdx.z;               // 0..5 -> 128 cols of I each
  const int beg = offs[e];
  const int cnt = offs[e + 1] - beg;
  if (tile * 16 >= cnt) return;
  const int pairBase = beg + tile * 16;
  const int rem = min(16, cnt - tile * 16);

  const int tid  = threadIdx.x;
  const int lane = tid & 31;
  const int wv   = tid >> 5;

  // A addressing: lane holds row (lane&15); K split 0/8 across lane halves
  const int lrow = lane & 15;
  const int ar   = (lrow < rem) ? lrow : rem - 1;
  const unsigned short* ap = xb + (size_t)pair_tok[pairBase + ar] * H_DIM;
  const int akb = (lane >> 4) * 8;

  // B addressing: lane holds column (lane&15); K split 0/8 across lane halves
  const int colG  = nz * 128 + wv * 16;
  const int bcol  = lane & 15;
  const int bkoff = (lane >> 4) * 8;
  const unsigned short* bgp = gupT + ((size_t)e * 2 * I_DIM + colG + bcol) * H_DIM;
  const unsigned short* bup = bgp + (size_t)I_DIM * H_DIM;

  v8f cg = {}; v8f cu = {};

  for (int k0 = 0; k0 < H_DIM; k0 += 32) {
    AFrag a, bg, bu;
    a.q[0]  = *(const uint4*)(ap + k0 + akb);          // K = akb..akb+7
    a.q[1]  = *(const uint4*)(ap + k0 + 16 + akb);     // K = 16+akb..
    bg.q[0] = *(const uint4*)(bgp + k0 + bkoff);       // K = bkoff..bkoff+7
    bg.q[1] = *(const uint4*)(bgp + k0 + 16 + bkoff);  // K = 16+bkoff..
    bu.q[0] = *(const uint4*)(bup + k0 + bkoff);
    bu.q[1] = *(const uint4*)(bup + k0 + 16 + bkoff);

    cg = __builtin_amdgcn_wmma_f32_16x16x32_bf16(false, a.v, false, bg.v,
                                                 (short)0, cg, false, false);
    cu = __builtin_amdgcn_wmma_f32_16x16x32_bf16(false, a.v, false, bu.v,
                                                 (short)0, cu, false, false);
  }

  // epilogue: h = silu(g) * u, store bf16
  const int orow0 = (lane >> 4) * 8;
  const int ocol  = lane & 15;
#pragma unroll
  for (int r = 0; r < 8; ++r) {
    const int row = orow0 + r;
    if (row < rem) {
      const float g = cg[r], u = cu[r];
      const float h = (g / (1.f + __expf(-g))) * u;
      hbuf[(size_t)(pairBase + row) * I_DIM + colG + ocol] = f2bf(h);
    }
  }
}

// ---------------------------------------------------------------- GEMM2
// out += w * (h @ Wd); tile 16 pairs x 256 cols of H (8 waves x 2x16 cols)
// A: hbuf rows (bf16); B: downT (E,H,I) bf16, K-contiguous per column
__global__ __launch_bounds__(256)
void gemm2_kernel(const unsigned short* __restrict__ hbuf,
                  const unsigned short* __restrict__ downT,
                  const int* __restrict__ offs,
                  const int* __restrict__ pair_tok,
                  const float* __restrict__ pair_w,
                  float* __restrict__ out) {
  const int e    = blockIdx.x;
  const int tile = blockIdx.y;
  const int nz   = blockIdx.z;               // 0..3 -> 256 cols of H each
  const int beg = offs[e];
  const int cnt = offs[e + 1] - beg;
  if (tile * 16 >= cnt) return;
  const int pairBase = beg + tile * 16;
  const int rem = min(16, cnt - tile * 16);

  const int tid  = threadIdx.x;
  const int lane = tid & 31;
  const int wv   = tid >> 5;

  __shared__ int   sTok[16];
  __shared__ float sW[16];
  if (tid < 16) {
    const int p = pairBase + (tid < rem ? tid : rem - 1);
    sTok[tid] = pair_tok[p];
    sW[tid]   = pair_w[p];
  }
  __syncthreads();

  const int lrow = lane & 15;
  const int ar   = (lrow < rem) ? lrow : rem - 1;
  const unsigned short* ap = hbuf + (size_t)(pairBase + ar) * I_DIM;
  const int akb = (lane >> 4) * 8;

  const int n0    = nz * 256 + wv * 16;      // two column tiles: n0, n0+128
  const int bcol  = lane & 15;
  const int bkoff = (lane >> 4) * 8;
  const unsigned short* bp0 = downT + ((size_t)e * H_DIM + n0 + bcol) * I_DIM;
  const unsigned short* bp1 = bp0 + (size_t)128 * I_DIM;

  v8f c0 = {}; v8f c1 = {};
  for (int k0 = 0; k0 < I_DIM; k0 += 32) {
    AFrag a, b0, b1;
    a.q[0]  = *(const uint4*)(ap + k0 + akb);
    a.q[1]  = *(const uint4*)(ap + k0 + 16 + akb);
    b0.q[0] = *(const uint4*)(bp0 + k0 + bkoff);
    b0.q[1] = *(const uint4*)(bp0 + k0 + 16 + bkoff);
    b1.q[0] = *(const uint4*)(bp1 + k0 + bkoff);
    b1.q[1] = *(const uint4*)(bp1 + k0 + 16 + bkoff);

    c0 = __builtin_amdgcn_wmma_f32_16x16x32_bf16(false, a.v, false, b0.v,
                                                 (short)0, c0, false, false);
    c1 = __builtin_amdgcn_wmma_f32_16x16x32_bf16(false, a.v, false, b1.v,
                                                 (short)0, c1, false, false);
  }

  const int orow0 = (lane >> 4) * 8;
  const int ocol  = lane & 15;
#pragma unroll
  for (int r = 0; r < 8; ++r) {
    const int row = orow0 + r;
    if (row < rem) {
      // each out element receives exactly 2 contributions (top-2):
      // (0+a)+b == (0+b)+a bitwise -> deterministic despite atomics
      const float w = sW[row];
      float* orow = out + (size_t)sTok[row] * H_DIM;
      atomicAdd(&orow[n0 + ocol],        c0[r] * w);
      atomicAdd(&orow[n0 + 128 + ocol],  c1[r] * w);
    }
  }
}

// ---------------------------------------------------------------- launch
extern "C" void kernel_launch(void* const* d_in, const int* in_sizes, int n_in,
                              void* d_out, int out_size, void* d_ws, size_t ws_size,
                              hipStream_t stream) {
  (void)in_sizes; (void)n_in; (void)out_size; (void)ws_size;
  const float* x    = (const float*)d_in[0];   // (T,H)
  const float* gw   = (const float*)d_in[1];   // (H,E)
  const float* gup  = (const float*)d_in[2];   // (E,H,2I)
  const float* down = (const float*)d_in[3];   // (E,I,H)
  float* out = (float*)d_out;                  // (T,H)

  char* p = (char*)d_ws;
  auto alloc = [&](size_t bytes) -> char* {
    char* r = p; p += (bytes + 255) & ~(size_t)255; return r;
  };
  int*   topk_idx = (int*)  alloc((size_t)T_TOK * TOPK * sizeof(int));
  float* topk_w   = (float*)alloc((size_t)T_TOK * TOPK * sizeof(float));
  int*   cnt      = (int*)  alloc(E_NUM * sizeof(int));
  int*   cursor   = (int*)  alloc(E_NUM * sizeof(int));
  int*   offs     = (int*)  alloc((E_NUM + 1) * sizeof(int));
  int*   pair_tok = (int*)  alloc((size_t)NPAIR * sizeof(int));
  float* pair_w   = (float*)alloc((size_t)NPAIR * sizeof(float));
  unsigned short* xb    = (unsigned short*)alloc((size_t)T_TOK * H_DIM * 2);
  unsigned short* hbuf  = (unsigned short*)alloc((size_t)NPAIR * I_DIM * 2);
  unsigned short* gupT  = (unsigned short*)alloc((size_t)E_NUM * 2 * I_DIM * H_DIM * 2);
  unsigned short* downT = (unsigned short*)alloc((size_t)E_NUM * I_DIM * H_DIM * 2);

  init_meta_kernel<<<1, 64, 0, stream>>>(cnt);
  cvt_x_kernel<<<(T_TOK * H_DIM) / 256, 256, 0, stream>>>(x, xb);
  // gate_up: (H, 2I) per expert -> (2I, H) bf16
  cvt_transpose_kernel<<<dim3(H_DIM / 32, 2 * I_DIM / 32, E_NUM), 256, 0, stream>>>(
      gup, gupT, H_DIM, 2 * I_DIM);
  // down: (I, H) per expert -> (H, I) bf16
  cvt_transpose_kernel<<<dim3(I_DIM / 32, H_DIM / 32, E_NUM), 256, 0, stream>>>(
      down, downT, I_DIM, H_DIM);
  gate_topk_kernel<<<T_TOK / 8, 256, 0, stream>>>(x, gw, topk_idx, topk_w, cnt);
  scan_kernel<<<1, 32, 0, stream>>>(cnt, offs, cursor);
  build_kernel<<<(T_TOK + 255) / 256, 256, 0, stream>>>(topk_idx, topk_w, offs,
                                                        cursor, pair_tok, pair_w);
  zero_kernel<<<(T_TOK * H_DIM) / 256, 256, 0, stream>>>(out);
  gemm1_kernel<<<dim3(E_NUM, T_TOK / 16, I_DIM / 128), 256, 0, stream>>>(
      xb, gupT, offs, pair_tok, hbuf);
  gemm2_kernel<<<dim3(E_NUM, T_TOK / 16, H_DIM / 256), 256, 0, stream>>>(
      hbuf, downT, offs, pair_tok, pair_w, out);
}
